// GTModel_73924977099256
// MI455X (gfx1250) — matrix operand order repaired and checked by reading, not compiled, not run on hardware
//
#include <hip/hip_runtime.h>
#include <hip/hip_bf16.h>

// ---------------------------------------------------------------------------
// Types for CDNA5 WMMA (gfx1250, wave32)
// ---------------------------------------------------------------------------
typedef __attribute__((ext_vector_type(16))) __bf16 v16bf;
typedef __attribute__((ext_vector_type(8)))  float  v8f;
typedef unsigned short ushort_t;
typedef unsigned int   uint_t;

#define WMMA_BF16(A, B, C) \
    __builtin_amdgcn_wmma_f32_16x16x32_bf16(false, (A), false, (B), (short)0, (C), false, false)

// fp32 -> bf16 round-to-nearest-even (bit pattern as ushort)
static __device__ __forceinline__ ushort_t f2bf(float f) {
    uint_t u = __float_as_uint(f);
    uint_t r = u + 0x7fffu + ((u >> 16) & 1u);
    return (ushort_t)(r >> 16);
}

// Load a 16-element bf16 fragment as two 16-byte LDS chunks.
static __device__ __forceinline__ v16bf ld_frag(const ushort_t* p0, const ushort_t* p1) {
    union { uint4 u[2]; v16bf v; } r;
    r.u[0] = *(const uint4*)p0;
    r.u[1] = *(const uint4*)p1;
    return r.v;
}

static __device__ __forceinline__ float redmax16(float x) {
    #pragma unroll
    for (int m = 1; m < 16; m <<= 1) x = fmaxf(x, __shfl_xor(x, m, 32));
    return x;
}
static __device__ __forceinline__ float redsum16(float x) {
    #pragma unroll
    for (int m = 1; m < 16; m <<= 1) x += __shfl_xor(x, m, 32);
    return x;
}

// ---------------------------------------------------------------------------
// Constants
// ---------------------------------------------------------------------------
#define BB   32
#define SS   1024
#define LQ   128
#define LK   896
#define HH   256
#define NHH  8
#define DHH  32
#define EPSF 1e-5f

// ---------------------------------------------------------------------------
// Weight pre-pass: fp32 W[K x 256] -> bf16 Wt[256 x K] (transposed, row = n)
// One-time cost (~2 MB total); removes per-block re-conversion in the GEMMs
// and enables pure b128 staging of the B operand.
// ---------------------------------------------------------------------------
__global__ __launch_bounds__(256) void convw_kernel(const float* __restrict__ W,
                                                    ushort_t* __restrict__ Wt, int K)
{
    const int k = blockIdx.x;    // 0..K-1
    const int n = threadIdx.x;   // 0..255
    Wt[(long)n * K + k] = f2bf(W[(long)k * 256 + n]);
}

// ---------------------------------------------------------------------------
// Generic bf16 WMMA GEMM:  C[M x 256] = A[M x K] @ W[K x 256] + bias
//   amode 0: A fp32 row-major (Af, lda)
//   amode 1: A bf16 row-major (Ab, lda)
//   amode 2: A = cate embedding gather: row r, col k -> table[cate[r*4+k/256]*256 + k%256]
//   amode 3: A fp32, row r -> Af[rowmap[r]*lda + k]
// Wt: bf16, transposed [256][K].
// Block: 256 threads (8 waves), tile 64(M) x 256(N), K-step 32.
// ---------------------------------------------------------------------------
__global__ __launch_bounds__(256) void gemm_kernel(
    const float* __restrict__ Af, const ushort_t* __restrict__ Ab, int lda, int amode,
    const int* __restrict__ rowmap, const int* __restrict__ cate_idx,
    const float* __restrict__ table,
    const ushort_t* __restrict__ Wt, const float* __restrict__ bias,
    float* __restrict__ C, int M, int K)
{
    (void)M;
    const int N = 256;
    __shared__ __align__(16) ushort_t As[64 * 32];   //  4 KB, A tile [m][k]
    __shared__ __align__(16) ushort_t Ws[256 * 32];  // 16 KB, B tile [n][k]

    const int tid  = threadIdx.x;
    const int wave = tid >> 5;
    const int lane = tid & 31;
    const int half = lane >> 4;
    const int l16  = lane & 15;
    const int wm   = (wave & 3) * 16;   // wave's M sub-offset (0..48)
    const int wn   = (wave >> 2) * 128; // wave's N half     (0 or 128)
    const long mrow0 = (long)blockIdx.x * 64;

    const v8f vzero = {0.f, 0.f, 0.f, 0.f, 0.f, 0.f, 0.f, 0.f};
    v8f acc[8];
    #pragma unroll
    for (int j = 0; j < 8; ++j) acc[j] = vzero;

    // A staging assignment: thread -> (row r, 8 consecutive k at k0)
    const int r  = tid >> 2;        // 0..63
    const int k0 = (tid & 3) * 8;   // 0,8,16,24
    long grow = mrow0 + r;
    if (amode == 3) grow = rowmap[mrow0 + r];

    for (int kk = 0; kk < K; kk += 32) {
        __syncthreads();
        // ---- stage A tile (fp32 -> bf16 where needed), b128 global reads ----
        if (amode == 1) {
            *(uint4*)(As + r * 32 + k0) =
                *(const uint4*)(Ab + grow * (long)lda + kk + k0);
        } else if (amode == 2) {
            const int slot = (kk + k0) >> 8;
            const int col  = (kk + k0) & 255;
            const long e   = cate_idx[grow * 4 + slot];
            const float4* src = (const float4*)(table + e * 256 + col);
            float4 a = src[0], b = src[1];
            ushort_t* dst = As + r * 32 + k0;
            dst[0] = f2bf(a.x); dst[1] = f2bf(a.y); dst[2] = f2bf(a.z); dst[3] = f2bf(a.w);
            dst[4] = f2bf(b.x); dst[5] = f2bf(b.y); dst[6] = f2bf(b.z); dst[7] = f2bf(b.w);
        } else {
            const float4* src = (const float4*)(Af + grow * (long)lda + kk + k0);
            float4 a = src[0], b = src[1];
            ushort_t* dst = As + r * 32 + k0;
            dst[0] = f2bf(a.x); dst[1] = f2bf(a.y); dst[2] = f2bf(a.z); dst[3] = f2bf(a.w);
            dst[4] = f2bf(b.x); dst[5] = f2bf(b.y); dst[6] = f2bf(b.z); dst[7] = f2bf(b.w);
            // prefetch next K-tile of the streaming A operand (global_prefetch_b8)
            if (kk + 32 < K)
                __builtin_prefetch(Af + grow * (long)lda + kk + 32 + k0, 0, 1);
        }
        // ---- stage W tile: pure b128 copies from pre-transposed bf16 [n][K] ----
        {
            const uint4* src = (const uint4*)(Wt + (long)tid * K + kk);
            uint4* dst = (uint4*)(Ws + tid * 32);
            dst[0] = src[0]; dst[1] = src[1]; dst[2] = src[2]; dst[3] = src[3];
        }
        __syncthreads();

        // ---- A fragment (16x32, ISA layout): chunks at k=8*half and k=16+8*half
        const ushort_t* arow = As + (wm + l16) * 32 + 8 * half;
        v16bf afrag = ld_frag(arow, arow + 16);

        // ---- batch all 8 B fragments, then issue WMMAs back-to-back ----
        v16bf bfrag[8];
        #pragma unroll
        for (int j = 0; j < 8; ++j) {
            const ushort_t* brow = Ws + (wn + 16 * j + l16) * 32 + 8 * half;
            bfrag[j] = ld_frag(brow, brow + 16);
        }
        #pragma unroll
        for (int j = 0; j < 8; ++j)
            acc[j] = WMMA_BF16(afrag, bfrag[j], acc[j]);
    }

    // ---- epilogue: bias + store (C/D layout: row = v + 8*half, col = lane%16)
    #pragma unroll
    for (int j = 0; j < 8; ++j) {
        const int col = wn + 16 * j + l16;
        const float b = bias ? bias[col] : 0.f;
        #pragma unroll
        for (int v = 0; v < 8; ++v) {
            const long row = mrow0 + wm + v + 8 * half;
            C[row * N + col] = acc[j][v] + b;
        }
    }
}

// ---------------------------------------------------------------------------
// BatchNorm stats over cont (32768 x 8): stats[0..7]=mean, stats[8..15]=rstd
// ---------------------------------------------------------------------------
__global__ __launch_bounds__(256) void bn_stats_kernel(const float* __restrict__ cont,
                                                       float* __restrict__ stats)
{
    __shared__ float red[256];
    float sum[8], sq[8];
    #pragma unroll
    for (int j = 0; j < 8; ++j) { sum[j] = 0.f; sq[j] = 0.f; }
    for (int r0 = threadIdx.x; r0 < BB * SS; r0 += 256) {
        #pragma unroll
        for (int j = 0; j < 8; ++j) {
            float x = cont[(long)r0 * 8 + j];
            sum[j] += x; sq[j] += x * x;
        }
    }
    for (int j = 0; j < 8; ++j) {
        red[threadIdx.x] = sum[j]; __syncthreads();
        for (int s = 128; s > 0; s >>= 1) { if (threadIdx.x < s) red[threadIdx.x] += red[threadIdx.x + s]; __syncthreads(); }
        float mu = red[0] / (float)(BB * SS);
        __syncthreads();
        red[threadIdx.x] = sq[j]; __syncthreads();
        for (int s = 128; s > 0; s >>= 1) { if (threadIdx.x < s) red[threadIdx.x] += red[threadIdx.x + s]; __syncthreads(); }
        if (threadIdx.x == 0) {
            float var = red[0] / (float)(BB * SS) - mu * mu;
            stats[j]     = mu;
            stats[8 + j] = rsqrtf(var + EPSF);
        }
        __syncthreads();
    }
}

// ---------------------------------------------------------------------------
// cont projection (K=8, VALU): out[row] = bn(cont[row]) @ cont_W + cont_b
// ---------------------------------------------------------------------------
__global__ __launch_bounds__(256) void cont_kernel(const float* __restrict__ cont,
                                                   const float* __restrict__ stats,
                                                   const float* __restrict__ bn_g,
                                                   const float* __restrict__ bn_b,
                                                   const float* __restrict__ W,
                                                   const float* __restrict__ bias,
                                                   float* __restrict__ out)
{
    __shared__ float cn[8];
    const long row = blockIdx.x;
    const int t = threadIdx.x;
    if (t < 8)
        cn[t] = (cont[row * 8 + t] - stats[t]) * stats[8 + t] * bn_g[t] + bn_b[t];
    __syncthreads();
    float s = bias[t];
    #pragma unroll
    for (int j = 0; j < 8; ++j) s += cn[j] * W[j * 256 + t];
    out[row * 256 + t] = s;
}

// ---------------------------------------------------------------------------
// Build row maps for query / key gathers (mask positions via query_index)
// ---------------------------------------------------------------------------
__global__ __launch_bounds__(256) void build_idx_kernel(const int* __restrict__ qidx,
                                                        int* __restrict__ rmq,
                                                        int* __restrict__ rmkv)
{
    __shared__ unsigned char flag[SS];
    __shared__ int kpos[LK];
    const int t = threadIdx.x;
    for (int i = t; i < SS; i += 256) flag[i] = 0;
    __syncthreads();
    if (t < 128) flag[qidx[t]] = 1;
    __syncthreads();
    if (t == 0) {
        int c = 0;
        for (int s = 0; s < SS; ++s) if (!flag[s]) kpos[c++] = s;
    }
    __syncthreads();
    for (int i = t; i < BB * LQ; i += 256) {
        int b = i >> 7, q = i & 127;
        rmq[i] = b * SS + qidx[q];
    }
    for (int i = t; i < BB * LK; i += 256) {
        int b = i / LK, k = i % LK;
        rmkv[i] = b * SS + kpos[k];
    }
}

// ---------------------------------------------------------------------------
// Triple LN + leaky -> bf16 concat row (768) for comb GEMM input
// ---------------------------------------------------------------------------
__device__ __forceinline__ float block_red_sum(float x, float* red) {
    const int t = threadIdx.x;
    red[t] = x; __syncthreads();
    for (int s = 128; s > 0; s >>= 1) { if (t < s) red[t] += red[t + s]; __syncthreads(); }
    float r = red[0]; __syncthreads();
    return r;
}

__global__ __launch_bounds__(256) void ln3_kernel(
    const float* __restrict__ P0, const float* __restrict__ P1, const float* __restrict__ P2,
    const float* __restrict__ g0, const float* __restrict__ b0,
    const float* __restrict__ g1, const float* __restrict__ b1,
    const float* __restrict__ g2, const float* __restrict__ b2,
    ushort_t* __restrict__ acomb)
{
    __shared__ float red[256];
    const long row = blockIdx.x;
    const int t = threadIdx.x;
    const float* Ps[3] = { P0, P1, P2 };
    const float* gs[3] = { g0, g1, g2 };
    const float* bs[3] = { b0, b1, b2 };
    for (int seg = 0; seg < 3; ++seg) {
        float x  = Ps[seg][row * 256 + t];
        float mu = block_red_sum(x, red) * (1.f / 256.f);
        float xc = x - mu;
        float var = block_red_sum(xc * xc, red) * (1.f / 256.f);
        float y = xc * rsqrtf(var + EPSF) * gs[seg][t] + bs[seg][t];
        y = (y >= 0.f) ? y : 0.01f * y;   // leaky applied to concatenated seq
        acomb[row * 768 + seg * 256 + t] = f2bf(y);
    }
}

// ---------------------------------------------------------------------------
// Generic LN over 256: optional +pos_emb (s = row % S), optional leaky after
// ---------------------------------------------------------------------------
__global__ __launch_bounds__(256) void ln_kernel(const float* __restrict__ in,
                                                 const float* __restrict__ g,
                                                 const float* __restrict__ b,
                                                 const float* __restrict__ pos,
                                                 int leaky_after,
                                                 float* __restrict__ out)
{
    __shared__ float red[256];
    const long row = blockIdx.x;
    const int t = threadIdx.x;
    float x  = in[row * 256 + t];
    float mu = block_red_sum(x, red) * (1.f / 256.f);
    float xc = x - mu;
    float var = block_red_sum(xc * xc, red) * (1.f / 256.f);
    float y = xc * rsqrtf(var + EPSF) * g[t] + b[t];
    if (pos) y += pos[(row & (SS - 1)) * 256 + t];
    if (leaky_after) y = (y >= 0.f) ? y : 0.01f * y;
    out[row * 256 + t] = y;
}

// ---------------------------------------------------------------------------
// Fused flash attention: one block per (b,h). 8 waves x 16 query rows.
// Streams K/V in 32-key chunks through LDS; online softmax; WMMA everywhere.
// ---------------------------------------------------------------------------
__global__ __launch_bounds__(256) void attn_kernel(const float* __restrict__ Qm,
                                                   const float* __restrict__ Km,
                                                   const float* __restrict__ Vm,
                                                   const int* __restrict__ mask,
                                                   float* __restrict__ Om)
{
    const float scale = 0.1767766952966369f; // 1/sqrt(32)
    const int b = blockIdx.x >> 3;
    const int h = blockIdx.x & 7;
    const int tid  = threadIdx.x;
    const int wave = tid >> 5;
    const int lane = tid & 31;
    const int half = lane >> 4;
    const int l16  = lane & 15;

    __shared__ __align__(16) ushort_t Kc[32 * 32];      // [key][dh]
    __shared__ __align__(16) ushort_t Vc[32 * 32];      // [dh][key]  (transposed)
    __shared__ __align__(16) ushort_t Pt[8][16 * 32];   // per-wave P tile [row][key]

    // --- load this wave's Q fragment (16 q-rows x 32 dh) once, A layout ---
    const long qrow = (long)b * LQ + wave * 16 + l16;
    const float* qp = Qm + qrow * 256 + h * DHH;
    union { ushort_t s[16]; v16bf v; } qf;
    #pragma unroll
    for (int i = 0; i < 8; ++i) qf.s[i]     = f2bf(qp[8 * half + i]);
    #pragma unroll
    for (int i = 0; i < 8; ++i) qf.s[8 + i] = f2bf(qp[16 + 8 * half + i]);

    const v8f vzero = {0.f, 0.f, 0.f, 0.f, 0.f, 0.f, 0.f, 0.f};
    v8f O0 = vzero, O1 = vzero;
    float m8[8], l8[8];
    #pragma unroll
    for (int v = 0; v < 8; ++v) { m8[v] = -3.0e38f; l8[v] = 0.f; }

    const int kl = tid >> 3;         // 0..31 key within chunk
    const int d0 = (tid & 7) * 4;    // 4 dh per thread

    for (int ck = 0; ck < LK / 32; ++ck) {
        const int kb = ck * 32;
        // ---- stage K/V chunk into LDS (fp32 -> bf16) ----
        {
            const float* kp = Km + ((long)b * LK + kb + kl) * 256 + h * DHH + d0;
            const float* vp = Vm + ((long)b * LK + kb + kl) * 256 + h * DHH + d0;
            #pragma unroll
            for (int i = 0; i < 4; ++i) Kc[kl * 32 + d0 + i] = f2bf(kp[i]);
            #pragma unroll
            for (int i = 0; i < 4; ++i) Vc[(d0 + i) * 32 + kl] = f2bf(vp[i]);
            if (kb + 32 < LK) {
                __builtin_prefetch(kp + 32 * 256, 0, 1);
                __builtin_prefetch(vp + 32 * 256, 0, 1);
            }
        }
        __syncthreads();

        // ---- scores: S = Q(16x32) @ K^T(32x32) as two 16x16 WMMAs ----
        const ushort_t* kr0 = Kc + l16 * 32 + 8 * half;
        const ushort_t* kr1 = Kc + (16 + l16) * 32 + 8 * half;
        v8f s0 = WMMA_BF16(qf.v, ld_frag(kr0, kr0 + 16), vzero);
        v8f s1 = WMMA_BF16(qf.v, ld_frag(kr1, kr1 + 16), vzero);

        const int mk0 = mask[b * LK + kb + l16];
        const int mk1 = mask[b * LK + kb + 16 + l16];

        // ---- online softmax update (rows = v + 8*half; cols = lane dim) ----
        #pragma unroll
        for (int v = 0; v < 8; ++v) {
            float a0 = s0[v] * scale; if (mk0) a0 = -1.0e9f;
            float a1 = s1[v] * scale; if (mk1) a1 = -1.0e9f;
            float rmax = redmax16(fmaxf(a0, a1));
            float newm = fmaxf(m8[v], rmax);
            float f  = __expf(m8[v] - newm);
            float p0 = __expf(a0 - newm);
            float p1 = __expf(a1 - newm);
            float rsum = redsum16(p0 + p1);
            l8[v] = l8[v] * f + rsum;
            m8[v] = newm;
            O0[v] = O0[v] * f;
            O1[v] = O1[v] * f;
            Pt[wave][(v + 8 * half) * 32 + l16]      = f2bf(p0);
            Pt[wave][(v + 8 * half) * 32 + 16 + l16] = f2bf(p1);
        }

        // ---- O += P(16x32) @ V(32x32): reload P in A layout, two WMMAs ----
        const ushort_t* pr = Pt[wave] + l16 * 32 + 8 * half;
        v16bf pf = ld_frag(pr, pr + 16);
        const ushort_t* vr0 = Vc + l16 * 32 + 8 * half;
        const ushort_t* vr1 = Vc + (16 + l16) * 32 + 8 * half;
        v16bf vf0 = ld_frag(vr0, vr0 + 16);
        v16bf vf1 = ld_frag(vr1, vr1 + 16);
        O0 = WMMA_BF16(pf, vf0, O0);
        O1 = WMMA_BF16(pf, vf1, O1);
        __syncthreads();
    }

    // ---- finalize: O / l, scatter to Om (B*LQ, 256) ----
    #pragma unroll
    for (int v = 0; v < 8; ++v) {
        float inv = (l8[v] > 0.f) ? (1.f / l8[v]) : 0.f;
        const long row = (long)b * LQ + wave * 16 + v + 8 * half;
        Om[row * 256 + h * DHH + l16]      = O0[v] * inv;
        Om[row * 256 + h * DHH + 16 + l16] = O1[v] * inv;
    }
}

// ---------------------------------------------------------------------------
// Host launcher
// ---------------------------------------------------------------------------
extern "C" void kernel_launch(void* const* d_in, const int* in_sizes, int n_in,
                              void* d_out, int out_size, void* d_ws, size_t ws_size,
                              hipStream_t stream)
{
    (void)in_sizes; (void)n_in; (void)out_size; (void)ws_size;

    const float* node   = (const float*)d_in[0];
    const int*   cate   = (const int*)  d_in[1];
    const float* cont   = (const float*)d_in[2];
    const int*   qidx   = (const int*)  d_in[3];
    const int*   mask   = (const int*)  d_in[4];
    const float* table  = (const float*)d_in[5];
    const float* pos    = (const float*)d_in[6];
    const float* node_W = (const float*)d_in[7];
    const float* node_b = (const float*)d_in[8];
    const float* node_g = (const float*)d_in[9];
    const float* node_bb= (const float*)d_in[10];
    const float* cate_W = (const float*)d_in[11];
    const float* cate_b = (const float*)d_in[12];
    const float* cate_g = (const float*)d_in[13];
    const float* cate_bb= (const float*)d_in[14];
    const float* bn_g   = (const float*)d_in[15];
    const float* bn_b   = (const float*)d_in[16];
    const float* cont_W = (const float*)d_in[17];
    const float* cont_b = (const float*)d_in[18];
    const float* cont_g = (const float*)d_in[19];
    const float* cont_bb= (const float*)d_in[20];
    const float* comb_W = (const float*)d_in[21];
    const float* comb_b = (const float*)d_in[22];
    const float* comb_g = (const float*)d_in[23];
    const float* comb_bb= (const float*)d_in[24];
    const float* q_W    = (const float*)d_in[25];
    const float* q_b    = (const float*)d_in[26];
    const float* k_W    = (const float*)d_in[27];
    const float* k_b    = (const float*)d_in[28];
    const float* v_W    = (const float*)d_in[29];
    const float* v_b    = (const float*)d_in[30];
    const float* reg_W1 = (const float*)d_in[31];
    const float* reg_b1 = (const float*)d_in[32];
    const float* reg_g  = (const float*)d_in[33];
    const float* reg_bb = (const float*)d_in[34];
    const float* reg_W2 = (const float*)d_in[35];
    const float* reg_b2 = (const float*)d_in[36];

    // ---- workspace layout (aliased; total ~187 MB) ----
    char* ws = (char*)d_ws;
    size_t off = 0;
    auto alloc = [&](size_t bytes) { size_t o = off; off = (off + bytes + 255) & ~(size_t)255; return o; };
    const size_t MROWS = (size_t)BB * SS;           // 32768
    float* STATS = (float*)(ws + alloc(16 * 4));
    int*   RMQ   = (int*)  (ws + alloc((size_t)BB * LQ * 4));
    int*   RMKV  = (int*)  (ws + alloc((size_t)BB * LK * 4));
    // pre-converted bf16 transposed weights [256][K]
    ushort_t* WB_node = (ushort_t*)(ws + alloc(256 * 1024 * 2));
    ushort_t* WB_cate = (ushort_t*)(ws + alloc(256 * 1024 * 2));
    ushort_t* WB_comb = (ushort_t*)(ws + alloc(256 * 768 * 2));
    ushort_t* WB_q    = (ushort_t*)(ws + alloc(256 * 256 * 2));
    ushort_t* WB_k    = (ushort_t*)(ws + alloc(256 * 256 * 2));
    ushort_t* WB_v    = (ushort_t*)(ws + alloc(256 * 256 * 2));
    ushort_t* WB_r1   = (ushort_t*)(ws + alloc(256 * 256 * 2));
    ushort_t* WB_r2   = (ushort_t*)(ws + alloc(256 * 256 * 2));
    float* SEQ   = (float*)(ws + alloc(MROWS * 256 * 4));
    char*  P0c   =         ws + alloc(MROWS * 256 * 4);
    float* P0    = (float*)P0c;
    float* P1    = (float*)(ws + alloc(MROWS * 256 * 4));
    float* P2    = (float*)(ws + alloc(MROWS * 256 * 4));
    char*  ACc   =         ws + alloc(MROWS * 768 * 2);
    ushort_t* ACOMB = (ushort_t*)ACc;
    // aliases (lifetimes do not overlap):
    float* Qm = (float*)ACc;                               //  4.19 MB
    float* Km = (float*)(ACc + (size_t)BB * LQ * 256 * 4); // 29.36 MB
    float* Vm = P0;                                        // 29.36 MB
    float* Om = (float*)(P0c + (size_t)BB * LK * 256 * 4); //  4.19 MB
    float* R1 = P1;
    float* H1 = P2;
    float* CPRE = P0;

    // ---- weight pre-conversion (one-time, tiny) ----
    convw_kernel<<<1024, 256, 0, stream>>>(node_W, WB_node, 1024);
    convw_kernel<<<1024, 256, 0, stream>>>(cate_W, WB_cate, 1024);
    convw_kernel<<< 768, 256, 0, stream>>>(comb_W, WB_comb, 768);
    convw_kernel<<< 256, 256, 0, stream>>>(q_W,    WB_q,    256);
    convw_kernel<<< 256, 256, 0, stream>>>(k_W,    WB_k,    256);
    convw_kernel<<< 256, 256, 0, stream>>>(v_W,    WB_v,    256);
    convw_kernel<<< 256, 256, 0, stream>>>(reg_W1, WB_r1,   256);
    convw_kernel<<< 256, 256, 0, stream>>>(reg_W2, WB_r2,   256);

    // ---- pipeline ----
    bn_stats_kernel<<<1, 256, 0, stream>>>(cont, STATS);
    build_idx_kernel<<<1, 256, 0, stream>>>(qidx, RMQ, RMKV);
    cont_kernel<<<BB * SS, 256, 0, stream>>>(cont, STATS, bn_g, bn_b, cont_W, cont_b, P2);

    // node_e pre-LN  (M=32768, K=1024)
    gemm_kernel<<<512, 256, 0, stream>>>(node, nullptr, 1024, 0, nullptr, nullptr, nullptr,
                                         WB_node, node_b, P0, 32768, 1024);
    // cate_e pre-LN via embedding gather (M=32768, K=1024)
    gemm_kernel<<<512, 256, 0, stream>>>(nullptr, nullptr, 0, 2, nullptr, cate, table,
                                         WB_cate, cate_b, P1, 32768, 1024);
    // LN x3 + leaky -> bf16 concat (32768 x 768)
    ln3_kernel<<<BB * SS, 256, 0, stream>>>(P0, P1, P2, node_g, node_bb, cate_g, cate_bb,
                                            cont_g, cont_bb, ACOMB);
    // comb GEMM (bf16 A, K=768)
    gemm_kernel<<<512, 256, 0, stream>>>(nullptr, ACOMB, 768, 1, nullptr, nullptr, nullptr,
                                         WB_comb, comb_b, CPRE, 32768, 768);
    // LN + pos_emb -> SEQ
    ln_kernel<<<BB * SS, 256, 0, stream>>>(CPRE, comb_g, comb_bb, pos, 0, SEQ);

    // Q/K/V projections with row-gather (no materialized gather copies)
    gemm_kernel<<<64, 256, 0, stream>>>(SEQ, nullptr, 256, 3, RMQ, nullptr, nullptr,
                                        WB_q, q_b, Qm, BB * LQ, 256);
    gemm_kernel<<<448, 256, 0, stream>>>(SEQ, nullptr, 256, 3, RMKV, nullptr, nullptr,
                                         WB_k, k_b, Km, BB * LK, 256);
    gemm_kernel<<<448, 256, 0, stream>>>(SEQ, nullptr, 256, 3, RMKV, nullptr, nullptr,
                                         WB_v, v_b, Vm, BB * LK, 256);

    // fused flash attention -> Om
    attn_kernel<<<BB * NHH, 256, 0, stream>>>(Qm, Km, Vm, mask, Om);

    // regression head
    gemm_kernel<<<64, 256, 0, stream>>>(Om, nullptr, 256, 0, nullptr, nullptr, nullptr,
                                        WB_r1, reg_b1, R1, BB * LQ, 256);
    ln_kernel<<<BB * LQ, 256, 0, stream>>>(R1, reg_g, reg_bb, nullptr, 1, H1);
    gemm_kernel<<<64, 256, 0, stream>>>(H1, nullptr, 256, 0, nullptr, nullptr, nullptr,
                                        WB_r2, reg_b2, (float*)d_out, BB * LQ, 256);
}
